// PatchTST_33930241638829
// MI455X (gfx1250) — matrix operand order repaired and checked by reading, not compiled, not run on hardware
//
#include <hip/hip_runtime.h>
#include <hip/hip_bf16.h>

// ---------------- problem constants ----------------
#define BB 112
#define LL 128
#define DD 768
#define HH 12
#define EE 64
#define FF 3072
#define NLAYER 3
#define MM (BB * LL)          // 14336 tokens
#define BN_EPS 1e-5f

// ---------------- WMMA / TDM types ----------------
typedef __attribute__((ext_vector_type(16))) __bf16 v16bf;
typedef __attribute__((ext_vector_type(8)))  float  v8f;
typedef __attribute__((ext_vector_type(4)))  unsigned int v4u;
typedef __attribute__((ext_vector_type(8)))  int v8i;
typedef __attribute__((ext_vector_type(4)))  int v4i;

#if defined(__gfx1250__) && __has_builtin(__builtin_amdgcn_tensor_load_to_lds)
#define USE_TDM 1
#endif

union FragU { uint4 q[2]; v16bf v; };
union Vec8U { uint4 q; unsigned short s[8]; };

__device__ __forceinline__ unsigned short f2bf(float f) {
    unsigned int u = __float_as_uint(f);
    unsigned int r = u + 0x7FFFu + ((u >> 16) & 1u);   // round to nearest even
    return (unsigned short)(r >> 16);
}

// A-fragment (16x32, M x K), row-major source with stride ld (in shorts).
// ISA layout: lane m=l&15, half=l>>4; VGPR i holds K = (i>>2)*16 + half*8 + (i&3)*2 (+1)
__device__ __forceinline__ v16bf load_a_frag(const unsigned short* base, int ld, int lane) {
    int m = lane & 15, half = lane >> 4;
    const unsigned short* r = base + m * ld + half * 8;
    FragU f;
    f.q[0] = *(const uint4*)(r);
    f.q[1] = *(const uint4*)(r + 16);
    return f.v;
}

// B-fragment (32x16, K x N) where B[k][n] = Wlds[n][k] (weights row-major [N,K]).
// ISA layout: lane n=l&15, half=l>>4; VGPR j holds K = (j>>2)*16 + half*8 + (j&3)*2 (+1)
__device__ __forceinline__ v16bf load_b_frag(const unsigned short* base, int ld, int lane) {
    int n = lane & 15, half = lane >> 4;
    const unsigned short* r = base + n * ld + half * 8;
    FragU f;
    f.q[0] = *(const uint4*)(r);
    f.q[1] = *(const uint4*)(r + 16);
    return f.v;
}

__device__ __forceinline__ v8f wmma_bf16(v16bf a, v16bf b, v8f c) {
    return __builtin_amdgcn_wmma_f32_16x16x32_bf16(false, a, false, b, (short)0, c, false, false);
}

#ifdef USE_TDM
// TDM load of a 128-row x 64-col bf16 tile (row stride K elements) into LDS with
// 16-short row padding (LDS row stride 80 shorts = 160B). D# per CDNA5 ISA ch.8:
//   pad_interval code 4 = 32 DWORDs (=128B = one 64-bf16 row), pad_amount code 7 = 8 DWORDs (=32B)
// This toolchain exposes the 6-arg builtin: (v4u g0, v8i g1, v4i g2, v4i g3, v8i extra, i32 cpol).
__device__ __forceinline__ void tdm_load_tile(const unsigned short* gsrc,
                                              const unsigned short* ldst, int K) {
    unsigned long long ga = (unsigned long long)(size_t)gsrc;
    unsigned lds_off = (unsigned)(size_t)(const void*)ldst;   // generic addr low 32 = LDS byte offset
    v4u g0;
    g0[0] = 1u;                                               // count=1 (valid), user mode
    g0[1] = lds_off;                                          // lds_addr (bytes)
    g0[2] = (unsigned)(ga & 0xFFFFFFFFu);                     // global_addr[31:0]
    g0[3] = (unsigned)((ga >> 32) & 0x01FFFFFFu) | (2u << 30); // global_addr[56:32] | type=2
    const unsigned td0 = 1u << 20, td1 = 1u << 20;            // tensor dims: large (no OOB in-tile)
    v8i g1;
    g1[0] = (int)((1u << 16) | (1u << 20) | (4u << 22) | (7u << 25)); // data_size=2B, pad_en, intv, amt
    g1[1] = (int)((td0 & 0xFFFFu) << 16);                     // [47:32]=ab_addr 0, [63:48]=td0.lo
    g1[2] = (int)(((td0 >> 16) & 0xFFFFu) | ((td1 & 0xFFFFu) << 16));
    g1[3] = (int)(((td1 >> 16) & 0xFFFFu) | (64u << 16));     // tile_dim0 = 64 elements
    g1[4] = (int)128;                                         // tile_dim1 = 128 rows, tile_dim2 = 0
    unsigned long long s0 = (unsigned long long)(unsigned)K;  // tensor_dim0_stride (elements)
    g1[5] = (int)(s0 & 0xFFFFFFFFu);
    g1[6] = (int)((s0 >> 32) & 0xFFFFu);                      // stride hi16 | dim1_stride.lo16 = 0
    g1[7] = 0;
    v4i gz; gz[0] = gz[1] = gz[2] = gz[3] = 0;                // 2D tensor: groups 2/3 zero
    v8i ge; ge[0] = ge[1] = ge[2] = ge[3] = ge[4] = ge[5] = ge[6] = ge[7] = 0;
    __builtin_amdgcn_tensor_load_to_lds(g0, g1, gz, gz, ge, 0);
}
#endif

// ---------------- fp32 -> bf16 convert ----------------
__global__ void cvt_bf16_k(const float* __restrict__ in, unsigned short* __restrict__ out, size_t n) {
    size_t i = (size_t)blockIdx.x * blockDim.x + threadIdx.x;
    if (i < n) out[i] = f2bf(in[i]);
}

// ---------------- GEMM: C[M,N] = A[M,K] @ W[N,K]^T + bias ----------------
// MODE 0: f32 out. MODE 1: bf16 out with exact-erf GELU. MODE 2: bf16 out.
template<int MODE>
__global__ __launch_bounds__(256)
void gemm_bf16(const unsigned short* __restrict__ A, const unsigned short* __restrict__ W,
               const float* __restrict__ bias, float* __restrict__ outF,
               unsigned short* __restrict__ outB, int M, int N, int K) {
    __shared__ unsigned short As[2][128 * 80];
    __shared__ unsigned short Bs[2][128 * 80];

    const int t = threadIdx.x;
    const int lane = t & 31, wave = t >> 5;
    const int wm = wave >> 1, wn = wave & 1;           // 4x2 wave grid -> 32x64 per wave
    const int m0 = blockIdx.y * 128, n0 = blockIdx.x * 128;

    v8f acc[2][4];
#pragma unroll
    for (int i = 0; i < 2; ++i)
#pragma unroll
        for (int j = 0; j < 4; ++j)
#pragma unroll
            for (int e = 0; e < 8; ++e) acc[i][j][e] = 0.0f;

#ifdef USE_TDM
    // ---- Tensor Data Mover double-buffered pipeline (wave 0 drives the DMA) ----
    if (wave == 0) {
        tdm_load_tile(A + (size_t)m0 * K, As[0], K);
        tdm_load_tile(W + (size_t)n0 * K, Bs[0], K);
    }
    for (int k0 = 0; k0 < K; k0 += 64) {
        const int buf = (k0 >> 6) & 1;
        const bool has_next = (k0 + 64 < K);
        if (wave == 0) {
            if (has_next) {
                tdm_load_tile(A + (size_t)m0 * K + k0 + 64, As[buf ^ 1], K);
                tdm_load_tile(W + (size_t)n0 * K + k0 + 64, Bs[buf ^ 1], K);
                __builtin_amdgcn_s_wait_tensorcnt(2);   // current pair complete, next in flight
            } else {
                __builtin_amdgcn_s_wait_tensorcnt(0);
            }
        }
        __syncthreads();
#pragma unroll
        for (int kk = 0; kk < 64; kk += 32) {
            v16bf a0 = load_a_frag(&As[buf][(wm * 32 + 0) * 80 + kk], 80, lane);
            v16bf a1 = load_a_frag(&As[buf][(wm * 32 + 16) * 80 + kk], 80, lane);
#pragma unroll
            for (int j = 0; j < 4; ++j) {
                v16bf b = load_b_frag(&Bs[buf][(wn * 64 + j * 16) * 80 + kk], 80, lane);
                acc[0][j] = wmma_bf16(a0, b, acc[0][j]);
                acc[1][j] = wmma_bf16(a1, b, acc[1][j]);
            }
        }
        __syncthreads();            // all waves done with buf before TDM may overwrite it
    }
#else
    // ---- fallback: synchronous global->VGPR->LDS staging ----
    for (int k0 = 0; k0 < K; k0 += 64) {
#pragma unroll
        for (int i = 0; i < 4; ++i) {                  // A tile: 128x64 bf16 (1024 x 16B)
            int idx = t + i * 256;
            int row = idx >> 3, cv = idx & 7;
            uint4 d = *(const uint4*)(A + (size_t)(m0 + row) * K + k0 + cv * 8);
            *(uint4*)(&As[0][row * 80 + cv * 8]) = d;
        }
#pragma unroll
        for (int i = 0; i < 4; ++i) {                  // W tile: 128x64 bf16
            int idx = t + i * 256;
            int row = idx >> 3, cv = idx & 7;
            uint4 d = *(const uint4*)(W + (size_t)(n0 + row) * K + k0 + cv * 8);
            *(uint4*)(&Bs[0][row * 80 + cv * 8]) = d;
        }
        if (k0 + 64 < K)
            __builtin_prefetch(A + (size_t)(m0 + (t >> 1)) * K + k0 + 64, 0, 0);
        __syncthreads();
#pragma unroll
        for (int kk = 0; kk < 64; kk += 32) {
            v16bf a0 = load_a_frag(&As[0][(wm * 32 + 0) * 80 + kk], 80, lane);
            v16bf a1 = load_a_frag(&As[0][(wm * 32 + 16) * 80 + kk], 80, lane);
#pragma unroll
            for (int j = 0; j < 4; ++j) {
                v16bf b = load_b_frag(&Bs[0][(wn * 64 + j * 16) * 80 + kk], 80, lane);
                acc[0][j] = wmma_bf16(a0, b, acc[0][j]);
                acc[1][j] = wmma_bf16(a1, b, acc[1][j]);
            }
        }
        __syncthreads();
    }
#endif

    // epilogue: C/D layout -> lane holds col n=l&15, rows m = v + (l>>4)*8
    const int ln = lane & 15, lh = lane >> 4;
#pragma unroll
    for (int i = 0; i < 2; ++i)
#pragma unroll
        for (int j = 0; j < 4; ++j) {
            int gm = m0 + wm * 32 + i * 16 + lh * 8;
            int gn = n0 + wn * 64 + j * 16 + ln;
            float bv = bias[gn];
#pragma unroll
            for (int v = 0; v < 8; ++v) {
                float val = acc[i][j][v] + bv;
                size_t off = (size_t)(gm + v) * N + gn;
                if (MODE == 0) {
                    outF[off] = val;
                } else if (MODE == 1) {
                    float g = 0.5f * val * (1.0f + erff(val * 0.70710678118f));
                    outB[off] = f2bf(g);
                } else {
                    outB[off] = f2bf(val);
                }
            }
        }
}

// ---------------- attention: one block per (b, h, query-half of 64 rows) ----------------
__global__ __launch_bounds__(128)
void attn_kernel(const unsigned short* __restrict__ Q, const unsigned short* __restrict__ Kd,
                 const unsigned short* __restrict__ V, const float* __restrict__ bias,
                 unsigned short* __restrict__ ctx) {
    __shared__ unsigned short Qs[64 * 136];   // Q strip, later reused as bf16 probabilities P[64][128]
    __shared__ unsigned short Ks[128 * 72];   // K  [s][e]
    __shared__ unsigned short VsT[64 * 136];  // V^T [e][s]

    const int g = blockIdx.x;
    const int b = g / (HH * 2);
    const int r = g % (HH * 2);
    const int h = r >> 1;
    const int q0 = (r & 1) * 64;
    const int t = threadIdx.x, lane = t & 31, wave = t >> 5;

#pragma unroll
    for (int i = 0; i < 4; ++i) {            // Q: 64x64
        int idx = t + i * 128;
        int row = idx >> 3, cv = idx & 7;
        uint4 d = *(const uint4*)(Q + ((size_t)(b * LL + q0 + row)) * DD + h * EE + cv * 8);
        *(uint4*)(Qs + row * 136 + cv * 8) = d;
    }
#pragma unroll
    for (int i = 0; i < 8; ++i) {            // K: 128x64
        int idx = t + i * 128;
        int row = idx >> 3, cv = idx & 7;
        uint4 d = *(const uint4*)(Kd + ((size_t)(b * LL + row)) * DD + h * EE + cv * 8);
        *(uint4*)(Ks + row * 72 + cv * 8) = d;
    }
#pragma unroll
    for (int i = 0; i < 8; ++i) {            // V: 128x64, transposed into VsT[e][s]
        int idx = t + i * 128;
        int s = idx >> 3, cv = idx & 7;
        Vec8U u;
        u.q = *(const uint4*)(V + ((size_t)(b * LL + s)) * DD + h * EE + cv * 8);
#pragma unroll
        for (int e = 0; e < 8; ++e) VsT[(cv * 8 + e) * 136 + s] = u.s[e];
    }
    __syncthreads();

    // scores: each wave computes a 16x128 strip (rows q0 + wave*16 ..)
    v8f sacc[8];
#pragma unroll
    for (int j = 0; j < 8; ++j)
#pragma unroll
        for (int e = 0; e < 8; ++e) sacc[j][e] = 0.0f;

#pragma unroll
    for (int kk = 0; kk < 64; kk += 32) {
        v16bf a = load_a_frag(Qs + (wave * 16) * 136 + kk, 136, lane);
#pragma unroll
        for (int j = 0; j < 8; ++j) {
            v16bf bfr = load_b_frag(Ks + (j * 16) * 72 + kk, 72, lane);
            sacc[j] = wmma_bf16(a, bfr, sacc[j]);
        }
    }

    // softmax over 128 keys; lane holds col n, rows m = v + half*8 (row lives in one 16-lane half)
    const int ln = lane & 15, lh = lane >> 4;
    const float scale = 0.125f;              // 1/sqrt(64)
    float rmax[8], rsum[8];
#pragma unroll
    for (int v = 0; v < 8; ++v) { rmax[v] = -1e30f; rsum[v] = 0.0f; }

#pragma unroll
    for (int j = 0; j < 8; ++j)
#pragma unroll
        for (int v = 0; v < 8; ++v) {
            int m = q0 + wave * 16 + lh * 8 + v;
            int n = j * 16 + ln;
            float s = sacc[j][v] * scale +
                      bias[(((size_t)b * LL + m) * LL + n) * HH + h];
            sacc[j][v] = s;
            rmax[v] = fmaxf(rmax[v], s);
        }
#pragma unroll
    for (int v = 0; v < 8; ++v)
        for (int off = 1; off < 16; off <<= 1)
            rmax[v] = fmaxf(rmax[v], __shfl_xor(rmax[v], off, 16));
#pragma unroll
    for (int j = 0; j < 8; ++j)
#pragma unroll
        for (int v = 0; v < 8; ++v) {
            float p = __expf(sacc[j][v] - rmax[v]);
            sacc[j][v] = p;
            rsum[v] += p;
        }
#pragma unroll
    for (int v = 0; v < 8; ++v) {
        for (int off = 1; off < 16; off <<= 1)
            rsum[v] += __shfl_xor(rsum[v], off, 16);
        rsum[v] = 1.0f / rsum[v];
    }
    // write bf16 probabilities back into this wave's own Qs strip (no cross-wave hazard)
#pragma unroll
    for (int j = 0; j < 8; ++j)
#pragma unroll
        for (int v = 0; v < 8; ++v)
            Qs[(wave * 16 + lh * 8 + v) * 136 + j * 16 + ln] = f2bf(sacc[j][v] * rsum[v]);

    // context = P(16x128) @ V(128x64)
#pragma unroll
    for (int nt = 0; nt < 4; ++nt) {
        v8f o;
#pragma unroll
        for (int e = 0; e < 8; ++e) o[e] = 0.0f;
#pragma unroll
        for (int kc = 0; kc < 4; ++kc) {
            v16bf a = load_a_frag(Qs + (wave * 16) * 136 + kc * 32, 136, lane);
            v16bf bfr = load_b_frag(VsT + (nt * 16) * 136 + kc * 32, 136, lane);
            o = wmma_bf16(a, bfr, o);
        }
#pragma unroll
        for (int v = 0; v < 8; ++v) {
            int m = q0 + wave * 16 + lh * 8 + v;
            int e = nt * 16 + ln;
            ctx[((size_t)(b * LL + m)) * DD + h * EE + e] = f2bf(o[v]);
        }
    }
}

// ---------------- BatchNorm over (B,L) per channel ----------------
__global__ void bn_zero(float* __restrict__ stats) {
    int i = blockIdx.x * blockDim.x + threadIdx.x;
    if (i < 2 * DD) stats[i] = 0.0f;
}

__global__ __launch_bounds__(256)
void bn_partial(const float* __restrict__ x, const float* __restrict__ nx,
                float* __restrict__ stats) {
    int r0 = blockIdx.x * 64;
    int t = threadIdx.x;
    float s1[3] = {0.f, 0.f, 0.f}, s2[3] = {0.f, 0.f, 0.f};
    for (int r = 0; r < 64; ++r) {
        size_t base = (size_t)(r0 + r) * DD;
#pragma unroll
        for (int c3 = 0; c3 < 3; ++c3) {
            int c = t + c3 * 256;
            float v = x[base + c];
            if (nx) v += nx[base + c];
            s1[c3] += v;
            s2[c3] += v * v;
        }
    }
#pragma unroll
    for (int c3 = 0; c3 < 3; ++c3) {
        int c = t + c3 * 256;
        atomicAdd(&stats[c], s1[c3]);
        atomicAdd(&stats[DD + c], s2[c3]);
    }
}

template<int WRITE_BF>
__global__ __launch_bounds__(256)
void bn_apply(const float* __restrict__ x, const float* __restrict__ nx,
              const float* __restrict__ stats, const float* __restrict__ gamma,
              const float* __restrict__ beta, float* __restrict__ xout,
              unsigned short* __restrict__ xbout) {
    size_t i = (size_t)blockIdx.x * 256 + threadIdx.x;
    if (i >= (size_t)MM * DD) return;
    int c = (int)(i % DD);
    const float invN = 1.0f / (float)MM;
    float mean = stats[c] * invN;
    float var = stats[DD + c] * invN - mean * mean;
    float v = x[i];
    if (nx) v += nx[i];
    float y = (v - mean) * rsqrtf(var + BN_EPS) * gamma[c] + beta[c];
    xout[i] = y;
    if (WRITE_BF) xbout[i] = f2bf(y);
}

// ---------------- host-side orchestration ----------------
extern "C" void kernel_launch(void* const* d_in, const int* in_sizes, int n_in,
                              void* d_out, int out_size, void* d_ws, size_t ws_size,
                              hipStream_t stream) {
    (void)in_sizes; (void)n_in; (void)out_size; (void)ws_size;

    const float* x_in  = (const float*)d_in[0];
    const float* abias = (const float*)d_in[1];
    const float* Wq = (const float*)d_in[2];  const float* bq = (const float*)d_in[3];
    const float* Wk = (const float*)d_in[4];  const float* bk = (const float*)d_in[5];
    const float* Wv = (const float*)d_in[6];  const float* bv = (const float*)d_in[7];
    const float* Wo = (const float*)d_in[8];  const float* bo = (const float*)d_in[9];
    const float* W1 = (const float*)d_in[10]; const float* b1 = (const float*)d_in[11];
    const float* W2 = (const float*)d_in[12]; const float* b2 = (const float*)d_in[13];
    const float* g1 = (const float*)d_in[14]; const float* be1 = (const float*)d_in[15];
    const float* g2 = (const float*)d_in[16]; const float* be2 = (const float*)d_in[17];
    const float* gf = (const float*)d_in[18]; const float* bef = (const float*)d_in[19];

    // workspace carve-up (256B aligned slices)
    size_t off = 0;
    auto take = [&](size_t elems, size_t esz) -> void* {
        void* p = (char*)d_ws + off;
        off += ((elems * esz + 255) / 256) * 256;
        return p;
    };
    const size_t WDD = (size_t)NLAYER * DD * DD;
    const size_t WFD = (size_t)NLAYER * FF * DD;
    unsigned short* wqB = (unsigned short*)take(WDD, 2);
    unsigned short* wkB = (unsigned short*)take(WDD, 2);
    unsigned short* wvB = (unsigned short*)take(WDD, 2);
    unsigned short* woB = (unsigned short*)take(WDD, 2);
    unsigned short* w1B = (unsigned short*)take(WFD, 2);
    unsigned short* w2B = (unsigned short*)take(WFD, 2);
    unsigned short* xb   = (unsigned short*)take((size_t)MM * DD, 2);
    unsigned short* qb   = (unsigned short*)take((size_t)MM * DD, 2);
    unsigned short* kb   = (unsigned short*)take((size_t)MM * DD, 2);
    unsigned short* vb   = (unsigned short*)take((size_t)MM * DD, 2);
    unsigned short* ctxb = (unsigned short*)take((size_t)MM * DD, 2);
    unsigned short* hb   = (unsigned short*)take((size_t)MM * FF, 2);
    float* nx    = (float*)take((size_t)MM * DD, 4);
    float* xcur  = (float*)take((size_t)MM * DD, 4);
    float* stats = (float*)take(2 * DD, 4);

    auto cvt = [&](const float* src, unsigned short* dst, size_t n) {
        cvt_bf16_k<<<(unsigned)((n + 255) / 256), 256, 0, stream>>>(src, dst, n);
    };

    // convert all weights to bf16 once
    cvt(Wq, wqB, WDD); cvt(Wk, wkB, WDD); cvt(Wv, wvB, WDD);
    cvt(Wo, woB, WDD); cvt(W1, w1B, WFD); cvt(W2, w2B, WFD);
    // initial activation copy to bf16
    cvt(x_in, xb, (size_t)MM * DD);

    const dim3 gD(DD / 128, MM / 128);   // (6,112)
    const dim3 gF(FF / 128, MM / 128);   // (24,112)
    const unsigned attnBlocks = BB * HH * 2;          // 2688
    const unsigned bnApplyBlocks = (unsigned)(((size_t)MM * DD + 255) / 256);

    const float* xprev = x_in;
    for (int l = 0; l < NLAYER; ++l) {
        // Q/K/V projections -> bf16
        gemm_bf16<2><<<gD, 256, 0, stream>>>(xb, wqB + (size_t)l * DD * DD, bq + l * DD,
                                             nullptr, qb, MM, DD, DD);
        gemm_bf16<2><<<gD, 256, 0, stream>>>(xb, wkB + (size_t)l * DD * DD, bk + l * DD,
                                             nullptr, kb, MM, DD, DD);
        gemm_bf16<2><<<gD, 256, 0, stream>>>(xb, wvB + (size_t)l * DD * DD, bv + l * DD,
                                             nullptr, vb, MM, DD, DD);
        // attention
        attn_kernel<<<attnBlocks, 128, 0, stream>>>(qb, kb, vb, abias, ctxb);
        // output projection -> f32
        gemm_bf16<0><<<gD, 256, 0, stream>>>(ctxb, woB + (size_t)l * DD * DD, bo + l * DD,
                                             nx, nullptr, MM, DD, DD);
        // BN(x + attn_out)
        bn_zero<<<(2 * DD + 255) / 256, 256, 0, stream>>>(stats);
        bn_partial<<<MM / 64, 256, 0, stream>>>(xprev, nx, stats);
        bn_apply<1><<<bnApplyBlocks, 256, 0, stream>>>(xprev, nx, stats, g1 + l * DD,
                                                       be1 + l * DD, xcur, xb);
        xprev = xcur;
        // FFN
        gemm_bf16<1><<<gF, 256, 0, stream>>>(xb, w1B + (size_t)l * FF * DD, b1 + l * FF,
                                             nullptr, hb, MM, FF, DD);
        gemm_bf16<0><<<gD, 256, 0, stream>>>(hb, w2B + (size_t)l * DD * FF, b2 + l * DD,
                                             nx, nullptr, MM, DD, FF);
        // BN(x + ffn_out)
        bn_zero<<<(2 * DD + 255) / 256, 256, 0, stream>>>(stats);
        bn_partial<<<MM / 64, 256, 0, stream>>>(xprev, nx, stats);
        bn_apply<1><<<bnApplyBlocks, 256, 0, stream>>>(xprev, nx, stats, g2 + l * DD,
                                                       be2 + l * DD, xcur, xb);
    }

    // final BN -> d_out (f32)
    bn_zero<<<(2 * DD + 255) / 256, 256, 0, stream>>>(stats);
    bn_partial<<<MM / 64, 256, 0, stream>>>(xcur, nullptr, stats);
    bn_apply<0><<<bnApplyBlocks, 256, 0, stream>>>(xcur, nullptr, stats, gf, bef,
                                                   (float*)d_out, nullptr);
}
